// PointerGeneratorLM_17265768530348
// MI455X (gfx1250) — matrix operand order repaired and checked by reading, not compiled, not run on hardware
//
#include <hip/hip_runtime.h>
#include <stdint.h>
#include <math.h>

#define B_  2
#define S_  1024
#define D_  512
#define V_  32000
#define TD_ 1536   /* 3*D */
#define CD_ 1024   /* 2*D */

typedef unsigned short u16;
typedef __attribute__((ext_vector_type(16))) __bf16 v16bf;
typedef __attribute__((ext_vector_type(8)))  float  v8f;

struct Frag32 { uint4 lo, hi; };
static_assert(sizeof(Frag32) == 32, "frag size");

__device__ __forceinline__ u16 f2bf(float f) {
  unsigned u = __float_as_uint(f);
  u += 0x7FFFu + ((u >> 16) & 1u);          // round-to-nearest-even
  return (u16)(u >> 16);
}
__device__ __forceinline__ float sigm(float x) { return 1.0f / (1.0f + __expf(-x)); }

// ---------------------------------------------------------------------------
// Register-blocked WMMA GEMM: C[m,n] = scale * sum_k A[m,k]*B[n,k] (+ bias[n])
// A: M x K bf16 row-major, B: N x K bf16 row-major (C = A * B^T), C: M x N f32.
// Batched via grid.z with element strides sA/sB/sC.
// Block = 256 threads = 8 waves arranged 2(M) x 4(N).
// Each wave computes a 2x4 grid of 16x16 tiles (32 rows x 64 cols):
//   per K-step: 2 A-frags + 4 B-frags (12 x b128) feed 8 WMMAs.
// Block output tile = 64(M) x 256(N).
// causal=1 skips waves strictly above the diagonal (attention scores).
// ---------------------------------------------------------------------------
__global__ void gemm_bf16_wmma(const u16* __restrict__ A, const u16* __restrict__ Bm,
                               const float* __restrict__ bias, float* __restrict__ C,
                               int M, int N, int K,
                               long long sA, long long sB, long long sC,
                               float scale, int causal)
{
  const int z = blockIdx.z;
  A  += (long long)z * sA;
  Bm += (long long)z * sB;
  C  += (long long)z * sC;

  const int wave = threadIdx.x >> 5;
  const int lane = threadIdx.x & 31;
  const int wn = wave & 3;                   // 4 waves along N
  const int wm = wave >> 2;                  // 2 waves along M
  const int ntile0 = (blockIdx.x * 4 + wn) * 4;   // first of 4 n-tiles
  const int mtile0 = (blockIdx.y * 2 + wm) * 2;   // first of 2 m-tiles
  if (ntile0 * 16 >= N) return;                   // wave-uniform
  if (causal && ntile0 > mtile0 + 1) return;      // wave-uniform

  const int lo = lane & 15;
  const int hi = lane >> 4;

  // A fragment base: row = mtile0*16+lo; chunks at k+hi*8 and k+hi*8+16
  const u16* ap = A  + (long long)(mtile0 * 16 + lo) * K + hi * 8;
  // B fragment base: "row" n = ntile0*16+lo; chunks at k+hi*8 and k+hi*8+16
  const u16* bp = Bm + (long long)(ntile0 * 16 + lo) * K + hi * 8;
  const long long tstep = 16LL * K;          // one 16-row tile step

  v8f acc[2][4];
#pragma unroll
  for (int mi = 0; mi < 2; ++mi)
#pragma unroll
    for (int ni = 0; ni < 4; ++ni)
      acc[mi][ni] = {};

  for (int k = 0; k < K; k += 32) {
    Frag32 fa[2], fb[4];
#pragma unroll
    for (int mi = 0; mi < 2; ++mi) {
      fa[mi].lo = *(const uint4*)(ap + mi * tstep + k);
      fa[mi].hi = *(const uint4*)(ap + mi * tstep + k + 16);
    }
#pragma unroll
    for (int ni = 0; ni < 4; ++ni) {
      fb[ni].lo = *(const uint4*)(bp + ni * tstep + k);
      fb[ni].hi = *(const uint4*)(bp + ni * tstep + k + 16);
    }
#pragma unroll
    for (int mi = 0; mi < 2; ++mi) {
      v16bf va = __builtin_bit_cast(v16bf, fa[mi]);
#pragma unroll
      for (int ni = 0; ni < 4; ++ni) {
        v16bf vb = __builtin_bit_cast(v16bf, fb[ni]);
        acc[mi][ni] = __builtin_amdgcn_wmma_f32_16x16x32_bf16(
            /*neg_a=*/false, va, /*neg_b=*/false, vb,
            /*c_mod=*/(short)0, acc[mi][ni], /*reuse_a=*/false, /*reuse_b=*/false);
      }
    }
  }

#pragma unroll
  for (int mi = 0; mi < 2; ++mi) {
    const int row0 = (mtile0 + mi) * 16 + hi * 8;
#pragma unroll
    for (int ni = 0; ni < 4; ++ni) {
      const int col = (ntile0 + ni) * 16 + lo;
      const float bv = bias ? bias[col] : 0.0f;
#pragma unroll
      for (int i = 0; i < 8; ++i)
        C[(long long)(row0 + i) * N + col] = acc[mi][ni][i] * scale + bv;
    }
  }
}

// ---------------------------------------------------------------------------
// x_emb = token_embed[x] + pos_embed  -> bf16
// ---------------------------------------------------------------------------
__global__ void embed_kernel(const int* __restrict__ x, const float* __restrict__ tok,
                             const float* __restrict__ pos, u16* __restrict__ xe)
{
  long long i = (long long)blockIdx.x * blockDim.x + threadIdx.x;
  if (i >= (long long)B_ * S_ * D_) return;
  int d = (int)(i % D_);
  long long row = i / D_;
  int s = (int)(row & (S_ - 1));
  float v = tok[(long long)x[row] * D_ + d] + pos[(long long)s * D_ + d];
  xe[i] = f2bf(v);
}

__global__ void to_bf16(const float* __restrict__ in, u16* __restrict__ out, long long n)
{
  long long i = (long long)blockIdx.x * blockDim.x + threadIdx.x;
  if (i >= n) return;
  out[i] = f2bf(in[i]);
}

// gru f32 -> bf16 (row-major) + bf16 transposed per batch ([b][d][s])
__global__ void gru_to_bf16(const float* __restrict__ gru, u16* __restrict__ gbf,
                            u16* __restrict__ gbfT)
{
  long long i = (long long)blockIdx.x * blockDim.x + threadIdx.x;
  if (i >= (long long)B_ * S_ * D_) return;
  int d = (int)(i % D_);
  long long row = i / D_;
  int b = (int)(row >> 10), s = (int)(row & (S_ - 1));
  u16 v = f2bf(gru[i]);
  gbf[i] = v;
  gbfT[(long long)b * D_ * S_ + (long long)d * S_ + s] = v;
}

// ---------------------------------------------------------------------------
// GRU scan: one block per batch, 512 threads (thread d owns hidden dim d).
// h lives in LDS; W_hh (3 MB) streams from L2 each step.
// ---------------------------------------------------------------------------
__global__ void gru_kernel(const float* __restrict__ xproj, const float* __restrict__ W_hh,
                           const float* __restrict__ b_hh, float* __restrict__ gru_out)
{
  __shared__ float h[D_];
  const int b = blockIdx.x, d = threadIdx.x;
  h[d] = 0.0f;
  const float4* wr = (const float4*)(W_hh + (long long)d * D_);
  const float4* wz = (const float4*)(W_hh + (long long)(D_ + d) * D_);
  const float4* wn = (const float4*)(W_hh + (long long)(2 * D_ + d) * D_);
  const float bhr = b_hh[d], bhz = b_hh[D_ + d], bhn = b_hh[2 * D_ + d];
  __syncthreads();
  for (int s = 0; s < S_; ++s) {
    float hr = bhr, hz = bhz, hn = bhn;
    const float4* hv4 = (const float4*)h;
#pragma unroll 4
    for (int j = 0; j < D_ / 4; ++j) {
      float4 hv = hv4[j], w1 = wr[j], w2 = wz[j], w3 = wn[j];
      hr += hv.x * w1.x + hv.y * w1.y + hv.z * w1.z + hv.w * w1.w;
      hz += hv.x * w2.x + hv.y * w2.y + hv.z * w2.z + hv.w * w2.w;
      hn += hv.x * w3.x + hv.y * w3.y + hv.z * w3.z + hv.w * w3.w;
    }
    const float* xp = xproj + ((long long)b * S_ + s) * TD_;
    const float r = sigm(xp[d] + hr);
    const float z = sigm(xp[D_ + d] + hz);
    const float n = tanhf(xp[2 * D_ + d] + r * hn);
    const float hold = h[d];
    const float hnew = (1.0f - z) * n + z * hold;
    __syncthreads();
    h[d] = hnew;
    gru_out[((long long)b * S_ + s) * D_ + d] = hnew;
    __syncthreads();
  }
}

// RoPE on Q and K (f32 in) -> bf16 out. One block per (b,s) row, 256 pairs.
__global__ void rope_kernel(const float* __restrict__ Q, const float* __restrict__ K,
                            u16* __restrict__ Qbf, u16* __restrict__ Kbf)
{
  const int row = blockIdx.x;               // b*S + s
  const int s  = row & (S_ - 1);
  const int d2 = threadIdx.x;               // 0..255
  const float freq = __powf(10000.0f, -(float)(2 * d2) / (float)D_);
  float sn, cs;
  __sincosf((float)s * freq, &sn, &cs);
  const long long base = (long long)row * D_ + 2 * d2;
  const float q0 = Q[base], q1 = Q[base + 1];
  Qbf[base]     = f2bf(q0 * cs - q1 * sn);
  Qbf[base + 1] = f2bf(q1 * cs + q0 * sn);
  const float k0 = K[base], k1 = K[base + 1];
  Kbf[base]     = f2bf(k0 * cs - k1 * sn);
  Kbf[base + 1] = f2bf(k1 * cs + k0 * sn);
}

// Causal softmax over scores row; writes probs f32 (in place) + bf16 copy.
__global__ void attn_softmax(float* __restrict__ scores, u16* __restrict__ probs_bf)
{
  __shared__ float red[256];
  const int row = blockIdx.x;               // b*S + i
  const int b = row >> 10, i = row & (S_ - 1);
  float* sr = scores + ((long long)b * S_ + i) * S_;
  const int t = threadIdx.x;

  float m = -3.0e38f;
  for (int j = t; j <= i; j += 256) m = fmaxf(m, sr[j]);
  red[t] = m; __syncthreads();
  for (int s = 128; s > 0; s >>= 1) { if (t < s) red[t] = fmaxf(red[t], red[t + s]); __syncthreads(); }
  m = red[0]; __syncthreads();

  float sum = 0.0f;
  for (int j = t; j <= i; j += 256) sum += __expf(sr[j] - m);
  red[t] = sum; __syncthreads();
  for (int s = 128; s > 0; s >>= 1) { if (t < s) red[t] += red[t + s]; __syncthreads(); }
  const float inv = 1.0f / red[0];

  for (int j = t; j < S_; j += 256) {
    float p = (j <= i) ? __expf(sr[j] - m) * inv : 0.0f;
    sr[j] = p;
    probs_bf[(long long)row * S_ + j] = f2bf(p);
  }
}

// combined = [gru_out, context] -> bf16
__global__ void concat_bf16(const float* __restrict__ gru, const float* __restrict__ ctx,
                            u16* __restrict__ comb, long long n)
{
  long long i = (long long)blockIdx.x * blockDim.x + threadIdx.x;
  if (i >= n) return;
  int d = (int)(i % CD_);
  long long row = i / CD_;
  float v = (d < D_) ? gru[row * D_ + d] : ctx[row * D_ + (d - D_)];
  comb[i] = f2bf(v);
}

// gate = sigmoid(combined . gate_W + gate_b), computed from f32 gru/ctx
__global__ void gate_kernel(const float* __restrict__ gru, const float* __restrict__ ctx,
                            const float* __restrict__ gate_W, const float* __restrict__ gate_b,
                            float* __restrict__ gate)
{
  __shared__ float red[256];
  const int row = blockIdx.x;
  const int t = threadIdx.x;
  float sum = 0.0f;
  for (int d = t; d < D_; d += 256)
    sum += gru[(long long)row * D_ + d] * gate_W[d]
         + ctx[(long long)row * D_ + d] * gate_W[D_ + d];
  red[t] = sum; __syncthreads();
  for (int s = 128; s > 0; s >>= 1) { if (t < s) red[t] += red[t + s]; __syncthreads(); }
  if (t == 0) gate[row] = sigm(red[0] + gate_b[0]);
}

// In-place over d_out: row softmax over V, then scale by gate[row].
__global__ void vocab_softmax_gate(float* __restrict__ out, const float* __restrict__ gate)
{
  __shared__ float red[1024];
  const int row = blockIdx.x;
  float* lr = out + (long long)row * V_;
  const int t = threadIdx.x;

  float m = -3.0e38f;
  for (int j = t; j < V_; j += 1024) m = fmaxf(m, lr[j]);
  red[t] = m; __syncthreads();
  for (int s = 512; s > 0; s >>= 1) { if (t < s) red[t] = fmaxf(red[t], red[t + s]); __syncthreads(); }
  m = red[0]; __syncthreads();

  float sum = 0.0f;
  for (int j = t; j < V_; j += 1024) sum += __expf(lr[j] - m);
  red[t] = sum; __syncthreads();
  for (int s = 512; s > 0; s >>= 1) { if (t < s) red[t] += red[t + s]; __syncthreads(); }
  const float g = gate[row] / red[0];

  for (int j = t; j < V_; j += 1024) lr[j] = g * __expf(lr[j] - m);
}

// out[b,i, x[b,j]] += (1-gate) * probs[b,i,j]   (causal j<=i), atomic f32 adds
__global__ void copy_scatter(const float* __restrict__ probs, const int* __restrict__ x,
                             const float* __restrict__ gate, float* __restrict__ out)
{
  const int row = blockIdx.x;               // b*S + i
  const int b = row >> 10, i = row & (S_ - 1);
  const float og = 1.0f - gate[row];
  const float* pr = probs + ((long long)b * S_ + i) * S_;
  float* orow = out + (long long)row * V_;
  for (int j = threadIdx.x; j <= i; j += 256)
    atomicAdd(orow + x[b * S_ + j], og * pr[j]);
}

// ---------------------------------------------------------------------------
extern "C" void kernel_launch(void* const* d_in, const int* in_sizes, int n_in,
                              void* d_out, int out_size, void* d_ws, size_t ws_size,
                              hipStream_t stream)
{
  (void)in_sizes; (void)n_in; (void)out_size; (void)ws_size;

  const int*   x      = (const int*)  d_in[0];
  const float* tok    = (const float*)d_in[1];
  const float* pos    = (const float*)d_in[2];
  const float* W_ih   = (const float*)d_in[3];
  const float* W_hh   = (const float*)d_in[4];
  const float* b_ih   = (const float*)d_in[5];
  const float* b_hh   = (const float*)d_in[6];
  const float* W_q    = (const float*)d_in[7];
  const float* W_k    = (const float*)d_in[8];
  const float* gen_W  = (const float*)d_in[9];
  const float* gen_b  = (const float*)d_in[10];
  const float* gate_W = (const float*)d_in[11];
  const float* gate_b = (const float*)d_in[12];
  float* out = (float*)d_out;

  size_t off = 0;
  auto carve = [&](size_t bytes) -> void* {
    void* p = (char*)d_ws + off;
    off += (bytes + 255) & ~(size_t)255;
    return p;
  };
  u16*   xe_bf    = (u16*)  carve((size_t)B_ * S_ * D_ * 2);
  u16*   Wih_bf   = (u16*)  carve((size_t)TD_ * D_ * 2);
  u16*   Wq_bf    = (u16*)  carve((size_t)D_ * D_ * 2);
  u16*   Wk_bf    = (u16*)  carve((size_t)D_ * D_ * 2);
  u16*   genW_bf  = (u16*)  carve((size_t)V_ * CD_ * 2);
  float* xproj    = (float*)carve((size_t)B_ * S_ * TD_ * 4);
  float* gru      = (float*)carve((size_t)B_ * S_ * D_ * 4);
  u16*   gru_bf   = (u16*)  carve((size_t)B_ * S_ * D_ * 2);
  u16*   gruT_bf  = (u16*)  carve((size_t)B_ * S_ * D_ * 2);
  float* Qf       = (float*)carve((size_t)B_ * S_ * D_ * 4);
  float* Kf       = (float*)carve((size_t)B_ * S_ * D_ * 4);
  u16*   Qbf      = (u16*)  carve((size_t)B_ * S_ * D_ * 2);
  u16*   Kbf      = (u16*)  carve((size_t)B_ * S_ * D_ * 2);
  float* scores   = (float*)carve((size_t)B_ * S_ * S_ * 4);
  u16*   probs_bf = (u16*)  carve((size_t)B_ * S_ * S_ * 2);
  float* ctx      = (float*)carve((size_t)B_ * S_ * D_ * 4);
  u16*   comb_bf  = (u16*)  carve((size_t)B_ * S_ * CD_ * 2);
  float* gate     = (float*)carve((size_t)B_ * S_ * 4);

  const int TPB = 256;
  auto nblk = [](long long n, int t) { return (unsigned)((n + t - 1) / t); };

  // 1) embeddings -> bf16
  embed_kernel<<<nblk((long long)B_ * S_ * D_, TPB), TPB, 0, stream>>>(x, tok, pos, xe_bf);

  // 2) weight conversions -> bf16
  to_bf16<<<nblk((long long)TD_ * D_, TPB), TPB, 0, stream>>>(W_ih, Wih_bf, (long long)TD_ * D_);
  to_bf16<<<nblk((long long)D_ * D_, TPB),  TPB, 0, stream>>>(W_q,  Wq_bf,  (long long)D_ * D_);
  to_bf16<<<nblk((long long)D_ * D_, TPB),  TPB, 0, stream>>>(W_k,  Wk_bf,  (long long)D_ * D_);
  to_bf16<<<nblk((long long)V_ * CD_, TPB), TPB, 0, stream>>>(gen_W, genW_bf, (long long)V_ * CD_);

  // Block tile = 64(M) x 256(N): grid.x = N/256, grid.y = M/64
  // 3) x_proj = x_emb @ W_ih^T + b_ih  (M=2048, N=1536, K=512)
  gemm_bf16_wmma<<<dim3(TD_ / 256, (B_ * S_) / 64, 1), 256, 0, stream>>>(
      xe_bf, Wih_bf, b_ih, xproj, B_ * S_, TD_, D_, 0, 0, 0, 1.0f, 0);

  // 4) GRU scan (sequential over S)
  gru_kernel<<<B_, D_, 0, stream>>>(xproj, W_hh, b_hh, gru);

  // 5) gru -> bf16 row-major + transposed
  gru_to_bf16<<<nblk((long long)B_ * S_ * D_, TPB), TPB, 0, stream>>>(gru, gru_bf, gruT_bf);

  // 6) Q/K projections (M=2048, N=512, K=512)
  gemm_bf16_wmma<<<dim3(D_ / 256, (B_ * S_) / 64, 1), 256, 0, stream>>>(
      gru_bf, Wq_bf, nullptr, Qf, B_ * S_, D_, D_, 0, 0, 0, 1.0f, 0);
  gemm_bf16_wmma<<<dim3(D_ / 256, (B_ * S_) / 64, 1), 256, 0, stream>>>(
      gru_bf, Wk_bf, nullptr, Kf, B_ * S_, D_, D_, 0, 0, 0, 1.0f, 0);

  // 7) RoPE -> bf16 Q/K
  rope_kernel<<<B_ * S_, D_ / 2, 0, stream>>>(Qf, Kf, Qbf, Kbf);

  // 8) scores = Q K^T / sqrt(D), batched, lower-triangular waves only
  gemm_bf16_wmma<<<dim3(S_ / 256, S_ / 64, B_), 256, 0, stream>>>(
      Qbf, Kbf, nullptr, scores, S_, S_, D_,
      (long long)S_ * D_, (long long)S_ * D_, (long long)S_ * S_,
      1.0f / sqrtf((float)D_), 1);

  // 9) causal softmax -> probs (f32 in place) + bf16
  attn_softmax<<<B_ * S_, 256, 0, stream>>>(scores, probs_bf);

  // 10) context = P @ gru_out  (B operand = gru^T, K-major)
  gemm_bf16_wmma<<<dim3(D_ / 256, S_ / 64, B_), 256, 0, stream>>>(
      probs_bf, gruT_bf, nullptr, ctx, S_, D_, S_,
      (long long)S_ * S_, (long long)D_ * S_, (long long)S_ * D_, 1.0f, 0);

  // 11) combined -> bf16
  concat_bf16<<<nblk((long long)B_ * S_ * CD_, TPB), TPB, 0, stream>>>(
      gru, ctx, comb_bf, (long long)B_ * S_ * CD_);

  // 12) gate
  gate_kernel<<<B_ * S_, 256, 0, stream>>>(gru, ctx, gate_W, gate_b, gate);

  // 13) logits = combined @ gen_W^T + gen_b  -> d_out  (M=2048, N=32000, K=1024)
  gemm_bf16_wmma<<<dim3(V_ / 256, (B_ * S_) / 64, 1), 256, 0, stream>>>(
      comb_bf, genW_bf, gen_b, out, B_ * S_, V_, CD_, 0, 0, 0, 1.0f, 0);

  // 14) P_gen softmax * gate, in place on d_out
  vocab_softmax_gate<<<B_ * S_, 1024, 0, stream>>>(out, gate);

  // 15) + (1-gate) * P_copy via atomic scatter
  copy_scatter<<<B_ * S_, 256, 0, stream>>>(scores, x, gate, out);
}
